// AdvancedSpatioTemporalGNN_71184787964031
// MI455X (gfx1250) — compile-verified, matrix operand order checked by reading, not run on hardware
//
#include <hip/hip_runtime.h>
#include <hip/hip_bf16.h>
#include <math.h>

// ---------------------------------------------------------------------------
// AdvancedSpatioTemporalGNN on MI455X (gfx1250).
// Dominant cost: diffusion GEMMs  S^T(512x512) @ X(512x128)  x 32 batches
// x 8 per cell x 48 cells ~= 0.9 TFLOP. Working set < L2 (192MB) -> compute
// bound -> v_wmma_f32_16x16x32_bf16 is the target path.
// ---------------------------------------------------------------------------

#define BB 32
#define TT 12
#define PP 12
#define NN 512
#define UU 64
#define LL 2

typedef __bf16 bf16_t;
typedef __bf16 v16bf __attribute__((ext_vector_type(16)));
typedef float  v8f   __attribute__((ext_vector_type(8)));

__device__ __forceinline__ v8f wmma_bf16(v16bf a, v16bf b, v8f c) {
    // D = A x B + C, f32 accumulate, bf16 operands (16x16x32)
    return __builtin_amdgcn_wmma_f32_16x16x32_bf16(false, a, false, b,
                                                   (short)0, c, false, false);
}

// 16-bit A fragment (16x32): lane<16 holds row=lane, K{0..7} and K{16..23};
// lane>=16 holds same row, K{8..15} and K{24..31}. Two 16B contiguous loads.
// B fragment mirrors this with lane = output column (operand pre-transposed
// in memory so per-lane K is contiguous).
__device__ __forceinline__ v16bf load_frag(const bf16_t* __restrict__ base,
                                           int row, int ld, int k0, int half) {
    union { float4 f4[2]; v16bf v; } u;
    const bf16_t* p = base + (size_t)row * ld + k0 + (half << 3);
    u.f4[0] = *reinterpret_cast<const float4*>(p);
    u.f4[1] = *reinterpret_cast<const float4*>(p + 16);
    return u.v;
}

// ---------------------------------------------------------------------------
// Propagation GEMM: Out[:, coloff:coloff+128] = ST @ XT^T  (per batch,support)
// ST: bf16 [4][512][512] row-major (already transposed+normalized supports)
// XT: bf16 channel-major [.. ][128][512]
// Out: bf16 [B][512][256] (fwd -> cols 0..127, bwd -> cols 128..255)
// grid = (2, 8, 4*B); block = 128 (4 waves, each a 32x32 output tile)
// ---------------------------------------------------------------------------
__global__ __launch_bounds__(128) void k_prop(const bf16_t* __restrict__ ST_all,
                                              const bf16_t* __restrict__ Xbase,
                                              bf16_t* __restrict__ Out1,
                                              bf16_t* __restrict__ Out2,
                                              int mode) {
    int z = blockIdx.z;
    int b = z >> 2, s = z & 3;
    const bf16_t* ST;
    const bf16_t* XT;
    bf16_t* Out;
    int coloff = (s & 1) * 128;
    if (mode == 0) {                       // xh: 4 supports on one X
        ST  = ST_all + (size_t)s * NN * NN;
        XT  = Xbase + (size_t)b * 128 * NN;
        Out = (s < 2 ? Out1 : Out2) + (size_t)b * NN * 256;
    } else {                               // xrh: branch 0 -> s1, branch 1 -> s2
        int br  = s >> 1;
        int sup = br * 2 + (s & 1);
        ST  = ST_all + (size_t)sup * NN * NN;
        XT  = Xbase + ((size_t)br * BB + b) * 128 * NN;
        Out = (br ? Out2 : Out1) + (size_t)b * NN * 256;
    }
    int wave = threadIdx.x >> 5;
    int lane = threadIdx.x & 31;
    int lrow = lane & 15, half = lane >> 4;
    int row0 = blockIdx.y * 64 + (wave >> 1) * 32;
    int c0   = blockIdx.x * 64 + (wave & 1) * 32;

    v8f z8 = {0.f, 0.f, 0.f, 0.f, 0.f, 0.f, 0.f, 0.f};
    v8f acc00 = z8, acc01 = z8, acc10 = z8, acc11 = z8;

    for (int k0 = 0; k0 < NN; k0 += 32) {
        if (k0 + 32 < NN) {  // gfx1250 global_prefetch_b8
            __builtin_prefetch(ST + (size_t)(row0 + lrow) * NN + k0 + 32, 0, 3);
            __builtin_prefetch(XT + (size_t)(c0 + lrow) * NN + k0 + 32, 0, 3);
        }
        v16bf a0 = load_frag(ST, row0 + lrow,      NN, k0, half);
        v16bf a1 = load_frag(ST, row0 + 16 + lrow, NN, k0, half);
        v16bf b0 = load_frag(XT, c0 + lrow,        NN, k0, half);
        v16bf b1 = load_frag(XT, c0 + 16 + lrow,   NN, k0, half);
        acc00 = wmma_bf16(a0, b0, acc00);
        acc01 = wmma_bf16(a0, b1, acc01);
        acc10 = wmma_bf16(a1, b0, acc10);
        acc11 = wmma_bf16(a1, b1, acc11);
    }
    // D layout: VGPR i, lane<16 -> (M=i, N=lane); lane>=16 -> (M=i+8)
    for (int i = 0; i < 8; ++i) {
        int m = half * 8 + i;
        Out[(size_t)(row0 + m)      * 256 + coloff + c0 + lrow]      = (bf16_t)acc00[i];
        Out[(size_t)(row0 + m)      * 256 + coloff + c0 + 16 + lrow] = (bf16_t)acc01[i];
        Out[(size_t)(row0 + 16 + m) * 256 + coloff + c0 + lrow]      = (bf16_t)acc10[i];
        Out[(size_t)(row0 + 16 + m) * 256 + coloff + c0 + 16 + lrow] = (bf16_t)acc11[i];
    }
}

// ---------------------------------------------------------------------------
// Gate GEMM + bias + activation:  out = act(P @ W + b)
// P: bf16 [B*512][256]; W: bf16 [od][256] (transposed); out: f32 [B*512][od]
// grid = (od/64, 256, 2); z picks (P1,W1,b1,o1) vs (P2,W2,b2,o2)
// act: 0 = sigmoid, 1 = tanh
// ---------------------------------------------------------------------------
__global__ __launch_bounds__(128) void k_gate(const bf16_t* __restrict__ P1,
                                              const bf16_t* __restrict__ P2,
                                              const bf16_t* __restrict__ W1,
                                              const bf16_t* __restrict__ W2,
                                              const float* __restrict__ b1,
                                              const float* __restrict__ b2,
                                              float* __restrict__ o1,
                                              float* __restrict__ o2,
                                              int od, int act) {
    const bf16_t* P    = blockIdx.z ? P2 : P1;
    const bf16_t* Wt   = blockIdx.z ? W2 : W1;
    const float*  bias = blockIdx.z ? b2 : b1;
    float*        outp = blockIdx.z ? o2 : o1;

    int wave = threadIdx.x >> 5;
    int lane = threadIdx.x & 31;
    int lrow = lane & 15, half = lane >> 4;
    int row0 = blockIdx.y * 64 + (wave >> 1) * 32;
    int c0   = blockIdx.x * 64 + (wave & 1) * 32;

    v8f z8 = {0.f, 0.f, 0.f, 0.f, 0.f, 0.f, 0.f, 0.f};
    v8f acc00 = z8, acc01 = z8, acc10 = z8, acc11 = z8;

    for (int k0 = 0; k0 < 256; k0 += 32) {
        v16bf a0 = load_frag(P, row0 + lrow,      256, k0, half);
        v16bf a1 = load_frag(P, row0 + 16 + lrow, 256, k0, half);
        v16bf w0 = load_frag(Wt, c0 + lrow,       256, k0, half);
        v16bf w1 = load_frag(Wt, c0 + 16 + lrow,  256, k0, half);
        acc00 = wmma_bf16(a0, w0, acc00);
        acc01 = wmma_bf16(a0, w1, acc01);
        acc10 = wmma_bf16(a1, w0, acc10);
        acc11 = wmma_bf16(a1, w1, acc11);
    }
    for (int i = 0; i < 8; ++i) {
        int m = half * 8 + i;
        int cols[2] = {c0 + lrow, c0 + 16 + lrow};
        int rows[2] = {row0 + m, row0 + 16 + m};
        float vals[2][2] = {{acc00[i], acc01[i]}, {acc10[i], acc11[i]}};
        for (int ti = 0; ti < 2; ++ti)
            for (int tj = 0; tj < 2; ++tj) {
                float v = vals[ti][tj] + bias[cols[tj]];
                v = act ? tanhf(v) : 1.0f / (1.0f + __expf(-v));
                outp[(size_t)rows[ti] * od + cols[tj]] = v;
            }
    }
}

// --------------------------- prep / elementwise ----------------------------

__global__ void k_zero(float* p, int n) {
    int i = blockIdx.x * 256 + threadIdx.x;
    if (i < n) p[i] = 0.0f;
}

// Stage 1: ah = adj*(tanh(0.5(sm+sm^T))+1) + I; row sums + (atomic) col sums
__global__ void k_prep_sup1(const float* __restrict__ adj,
                            const float* __restrict__ sm1,
                            const float* __restrict__ sm2,
                            float* __restrict__ AH,
                            float* __restrict__ sum_out,
                            float* __restrict__ sum_in) {
    int n = blockIdx.x, mat = blockIdx.y;
    const float* sm = mat ? sm2 : sm1;
    float* ah = AH + (size_t)mat * NN * NN;
    __shared__ float red[256];
    float rs = 0.f;
    for (int m = threadIdx.x; m < NN; m += 256) {
        float mask = tanhf(0.5f * (sm[n * NN + m] + sm[m * NN + n])) + 1.0f;
        float v = adj[n * NN + m] * mask + (m == n ? 1.0f : 0.0f);
        ah[(size_t)n * NN + m] = v;
        rs += v;
        atomicAdd(&sum_in[mat * NN + m], v);
    }
    red[threadIdx.x] = rs;
    __syncthreads();
    for (int s = 128; s > 0; s >>= 1) {
        if ((int)threadIdx.x < s) red[threadIdx.x] += red[threadIdx.x + s];
        __syncthreads();
    }
    if (threadIdx.x == 0) sum_out[mat * NN + n] = red[0];
}

// Stage 2: transposed normalized supports (bf16):
//   ST[2*mat+0][m][n] = ah[n][m]/d_out[n]   (fwd^T)
//   ST[2*mat+1][m][n] = ah[m][n]/d_in[n]    (bwd^T)
__global__ void k_prep_sup2(const float* __restrict__ AH,
                            const float* __restrict__ sum_out,
                            const float* __restrict__ sum_in,
                            bf16_t* __restrict__ STo) {
    int idx = blockIdx.x * 256 + threadIdx.x;   // 4*512*512
    int s = idx >> 18;
    int rem = idx & ((1 << 18) - 1);
    int m = rem >> 9, n = rem & (NN - 1);
    int mat = s >> 1, dir = s & 1;
    const float* ah = AH + (size_t)mat * NN * NN;
    float v;
    if (dir == 0) {
        float d = sum_out[mat * NN + n];
        v = ah[(size_t)n * NN + m] * (d > 0.f ? 1.f / d : 0.f);
    } else {
        float d = sum_in[mat * NN + n];
        v = ah[(size_t)m * NN + n] * (d > 0.f ? 1.f / d : 0.f);
    }
    STo[idx] = (bf16_t)v;
}

// Pre-sum (over K) and transpose weights: wt[l][o][k] (bf16), k<128 -> fwd sum
__global__ void k_prep_w(const float* __restrict__ w, bf16_t* __restrict__ wt,
                         int od) {
    int idx = blockIdx.x * 256 + threadIdx.x;
    if (idx >= 2 * od * 256) return;
    int l = idx / (od * 256);
    int rem = idx % (od * 256);
    int o = rem / 256, k = rem % 256;
    int j0 = (k < 128) ? 0 : 2;
    int c = k & 127;
    size_t base = ((size_t)(l * 4 + j0) * 128 + c) * od + o;
    wt[idx] = (bf16_t)(w[base] + w[base + (size_t)128 * od]);
}

// fc_in: xin[b][n][u] = src*(scale)*w[u]+b[u]; mode 0: x[:,t]*(tanh(tm_t)+1)
__global__ void k_fc_in(float* __restrict__ xin, const float* __restrict__ x,
                        const float* __restrict__ tm,
                        const float* __restrict__ w_in,
                        const float* __restrict__ b_in,
                        const float* __restrict__ go, int t, int mode) {
    int idx = blockIdx.x * 256 + threadIdx.x;   // B*512*64
    int u = idx & 63, n = (idx >> 6) & (NN - 1), b = idx >> 15;
    float v;
    if (mode == 0)
        v = x[((size_t)b * TT + t) * NN + n] * (tanhf(tm[t * NN + n]) + 1.0f);
    else
        v = go[b * NN + n];
    xin[idx] = v * w_in[u] + b_in[u];
}

// xh^T (channel-major bf16): c<64 -> xin, else h
__global__ void k_build_xhT(bf16_t* __restrict__ xht,
                            const float* __restrict__ xin,
                            const float* __restrict__ h) {
    int idx = blockIdx.x * 256 + threadIdx.x;   // B*128*512
    int n = idx & (NN - 1), c = (idx >> 9) & 127, b = idx >> 16;
    float v = (c < 64) ? xin[((size_t)b * NN + n) * 64 + c]
                       : h[((size_t)b * NN + n) * 64 + (c - 64)];
    xht[idx] = (bf16_t)v;
}

// [x, r*h]^T for both branches (r already sigmoid-ed in k_gate epilogue)
__global__ void k_build_xrhT(bf16_t* __restrict__ xrht,
                             const float* __restrict__ xin,
                             const float* __restrict__ h,
                             const float* __restrict__ ru1,
                             const float* __restrict__ ru2) {
    int idx = blockIdx.x * 256 + threadIdx.x;   // 2*B*128*512
    int n = idx & (NN - 1), c = (idx >> 9) & 127;
    int b = (idx >> 16) & (BB - 1), br = idx >> 21;
    float v;
    if (c < 64) {
        v = xin[((size_t)b * NN + n) * 64 + c];
    } else {
        const float* ru = br ? ru2 : ru1;
        float r = ru[((size_t)b * NN + n) * 128 + (c - 64)];
        v = r * h[((size_t)b * NN + n) * 64 + (c - 64)];
    }
    xrht[idx] = (bf16_t)v;
}

// h = (1-u)*h + u*c, u = 0.5(u1+u2), c = 0.5(c1+c2)
__global__ void k_update_h(float* __restrict__ h, const float* __restrict__ ru1,
                           const float* __restrict__ ru2,
                           const float* __restrict__ c1,
                           const float* __restrict__ c2) {
    int idx = blockIdx.x * 256 + threadIdx.x;   // B*512*64
    int u = idx & 63;
    size_t bn = (size_t)(idx >> 6);
    float um = 0.5f * (ru1[bn * 128 + 64 + u] + ru2[bn * 128 + 64 + u]);
    float cm = 0.5f * (c1[idx] + c2[idx]);
    h[idx] = (1.0f - um) * h[idx] + um * cm;
}

// fc_out: out[b][p][n] = h_last[b][n][:] . w + b; also feeds decoder go
__global__ void k_fc_out(const float* __restrict__ h,
                         const float* __restrict__ w,
                         const float* __restrict__ b, float* __restrict__ out,
                         float* __restrict__ go, int p) {
    int idx = blockIdx.x * 256 + threadIdx.x;   // B*512
    int n = idx & (NN - 1), bb = idx >> 9;
    float s = b[0];
    const float* hp = h + (size_t)idx * 64;
#pragma unroll
    for (int u = 0; u < 64; ++u) s += hp[u] * w[u];
    out[((size_t)bb * PP + p) * NN + n] = s;
    go[idx] = s;
}

// ---------------------------------------------------------------------------

extern "C" void kernel_launch(void* const* d_in, const int* in_sizes, int n_in,
                              void* d_out_v, int out_size, void* d_ws,
                              size_t ws_size, hipStream_t stream) {
    const float* x        = (const float*)d_in[0];
    const float* adj      = (const float*)d_in[1];
    const float* sm1      = (const float*)d_in[2];
    const float* sm2      = (const float*)d_in[3];
    const float* tm       = (const float*)d_in[4];
    const float* fc_in_w  = (const float*)d_in[5];
    const float* fc_in_b  = (const float*)d_in[6];
    const float* fc_out_w = (const float*)d_in[7];
    const float* fc_out_b = (const float*)d_in[8];
    const float* W[8];
    const float* Bs[8];
    for (int i = 0; i < 8; ++i) {           // enc_ru, enc_c, enc_ru2, enc_c2,
        W[i]  = (const float*)d_in[9 + 2 * i];   // dec_ru, dec_c, dec_ru2, dec_c2
        Bs[i] = (const float*)d_in[10 + 2 * i];
    }
    float* out = (float*)d_out_v;
    char* ws = (char*)d_ws;

    // ---- workspace layout (256B aligned regions) ----
    size_t off = 0;
    auto carve = [&](size_t bytes) {
        void* p = ws + off;
        off += (bytes + 255) & ~(size_t)255;
        return p;
    };
    float*  AH    = (float*)carve((size_t)2 * NN * NN * 4);
    float*  SOUT  = (float*)carve((size_t)2 * NN * 4);
    float*  SIN   = (float*)carve((size_t)2 * NN * 4);
    bf16_t* ST    = (bf16_t*)carve((size_t)4 * NN * NN * 2);
    bf16_t* WT    = (bf16_t*)carve((size_t)2 * 768 * 256 * 2);
    float*  H     = (float*)carve((size_t)LL * BB * NN * UU * 4);
    float*  XIN   = (float*)carve((size_t)BB * NN * UU * 4);
    bf16_t* XHT   = (bf16_t*)carve((size_t)BB * 128 * NN * 2);
    bf16_t* XRHT  = (bf16_t*)carve((size_t)2 * BB * 128 * NN * 2);
    bf16_t* P1    = (bf16_t*)carve((size_t)BB * NN * 256 * 2);
    bf16_t* P2    = (bf16_t*)carve((size_t)BB * NN * 256 * 2);
    bf16_t* Q1    = (bf16_t*)carve((size_t)BB * NN * 256 * 2);
    bf16_t* Q2    = (bf16_t*)carve((size_t)BB * NN * 256 * 2);
    float*  RU1   = (float*)carve((size_t)BB * NN * 128 * 4);
    float*  RU2   = (float*)carve((size_t)BB * NN * 128 * 4);
    float*  C1    = (float*)carve((size_t)BB * NN * UU * 4);
    float*  C2    = (float*)carve((size_t)BB * NN * UU * 4);
    float*  GO    = (float*)carve((size_t)BB * NN * 4);
    (void)ws_size;

    // WT per-tensor element offsets (layout [l][o][256])
    const int od_tab[8] = {128, 64, 128, 64, 128, 64, 128, 64};
    size_t wtoff[8];
    {
        size_t o = 0;
        for (int i = 0; i < 8; ++i) { wtoff[i] = o; o += (size_t)LL * od_tab[i] * 256; }
    }

    // ---- prep ----
    k_zero<<<dim3((2 * NN + 255) / 256), 256, 0, stream>>>(SIN, 2 * NN);
    k_zero<<<dim3((LL * BB * NN * UU + 255) / 256), 256, 0, stream>>>(H, LL * BB * NN * UU);
    k_zero<<<dim3((BB * NN + 255) / 256), 256, 0, stream>>>(GO, BB * NN);
    k_prep_sup1<<<dim3(NN, 2), 256, 0, stream>>>(adj, sm1, sm2, AH, SOUT, SIN);
    k_prep_sup2<<<dim3(4 * NN * NN / 256), 256, 0, stream>>>(AH, SOUT, SIN, ST);
    for (int i = 0; i < 8; ++i)
        k_prep_w<<<dim3((2 * od_tab[i] * 256 + 255) / 256), 256, 0, stream>>>(
            W[i], WT + wtoff[i], od_tab[i]);

    // ---- one DCGRU cell (dual-support) ----
    auto cell = [&](const float* xin_ptr, float* h_ptr, int wbase, int l) {
        bf16_t* WTru1 = WT + wtoff[wbase + 0] + (size_t)l * 128 * 256;
        bf16_t* WTc1  = WT + wtoff[wbase + 1] + (size_t)l * 64 * 256;
        bf16_t* WTru2 = WT + wtoff[wbase + 2] + (size_t)l * 128 * 256;
        bf16_t* WTc2  = WT + wtoff[wbase + 3] + (size_t)l * 64 * 256;
        const float* bru1 = Bs[wbase + 0] + l * 128;
        const float* bc1  = Bs[wbase + 1] + l * 64;
        const float* bru2 = Bs[wbase + 2] + l * 128;
        const float* bc2  = Bs[wbase + 3] + l * 64;

        k_build_xhT<<<dim3(BB * 128 * NN / 256), 256, 0, stream>>>(XHT, xin_ptr, h_ptr);
        k_prop<<<dim3(2, 8, 4 * BB), 128, 0, stream>>>(ST, XHT, P1, P2, 0);
        k_gate<<<dim3(2, BB * NN / 64, 2), 128, 0, stream>>>(
            P1, P2, WTru1, WTru2, bru1, bru2, RU1, RU2, 128, /*sigmoid*/ 0);
        k_build_xrhT<<<dim3(2 * BB * 128 * NN / 256), 256, 0, stream>>>(
            XRHT, xin_ptr, h_ptr, RU1, RU2);
        k_prop<<<dim3(2, 8, 4 * BB), 128, 0, stream>>>(ST, XRHT, Q1, Q2, 1);
        k_gate<<<dim3(1, BB * NN / 64, 2), 128, 0, stream>>>(
            Q1, Q2, WTc1, WTc2, bc1, bc2, C1, C2, 64, /*tanh*/ 1);
        k_update_h<<<dim3(BB * NN * UU / 256), 256, 0, stream>>>(h_ptr, RU1, RU2, C1, C2);
    };

    float* H0 = H;
    float* H1 = H + (size_t)BB * NN * UU;

    // ---- encoder ----
    for (int t = 0; t < TT; ++t) {
        k_fc_in<<<dim3(BB * NN * UU / 256), 256, 0, stream>>>(
            XIN, x, tm, fc_in_w, fc_in_b, GO, t, /*enc*/ 0);
        cell(XIN, H0, /*enc*/ 0, 0);
        cell(H0, H1, /*enc*/ 0, 1);
    }
    // ---- decoder ----
    for (int p = 0; p < PP; ++p) {
        k_fc_in<<<dim3(BB * NN * UU / 256), 256, 0, stream>>>(
            XIN, x, tm, fc_in_w, fc_in_b, GO, 0, /*dec*/ 1);
        cell(XIN, H0, /*dec*/ 4, 0);
        cell(H0, H1, /*dec*/ 4, 1);
        k_fc_out<<<dim3(BB * NN / 256), 256, 0, stream>>>(
            H1, fc_out_w, fc_out_b, out, GO, p);
    }
}
